// GraphAttention_77601469104331
// MI455X (gfx1250) — matrix operand order repaired and checked by reading, not compiled, not run on hardware
//
#include <hip/hip_runtime.h>
#include <hip/hip_bf16.h>

#define NEG_SLOPE 0.2f

typedef __attribute__((ext_vector_type(16))) __bf16 v16bf;
typedef __attribute__((ext_vector_type(8)))  float  v8f;

// -------------------------------------------------------------------------
// Kernel 1: th = x @ W^T  via v_wmma_f32_16x16x32_bf16.
// Block = 256 threads (8 waves), each block does a 128-row strip of x.
// Wave w owns rows [blk*128 + w*16, +16), loops over 8 column tiles of 16.
// W (128x128 f32) is staged to LDS as bf16 once per block.
// -------------------------------------------------------------------------
__global__ __launch_bounds__(256) void gat_gemm_th(
    const float* __restrict__ x, const float* __restrict__ W,
    float* __restrict__ th, int N)
{
    __shared__ __bf16 Wlds[128 * 128];   // 32 KB, [o][k] row-major

    const int tid = threadIdx.x;
    // Stage W -> LDS (f32 -> bf16), float4 loads
    for (int i = tid; i < (128 * 128) / 4; i += 256) {
        float4 w4 = ((const float4*)W)[i];
        int b = i * 4;
        Wlds[b + 0] = (__bf16)w4.x;
        Wlds[b + 1] = (__bf16)w4.y;
        Wlds[b + 2] = (__bf16)w4.z;
        Wlds[b + 3] = (__bf16)w4.w;
    }
    __syncthreads();

    const int wave   = tid >> 5;
    const int lane   = tid & 31;
    const int laneHi = lane >> 4;     // 0 or 1
    const int lane16 = lane & 15;
    const int row0   = blockIdx.x * 128 + wave * 16;

    // Wave-uniform exit: this wave's 16 rows are entirely out of range.
    // (No barriers below, so early return is safe.)
    if (row0 >= N) return;
    const bool fullWave = (row0 + 16 <= N);

    // ---- A fragments: rows of x, bf16, 4 k-steps of 32 ----
    // A 16x32, contiguous k order chosen here: lane row M = lane&15; element i -> K = i + 16*laneHi
    const int arow    = row0 + lane16;
    const float* xrow = x + (size_t)((arow < N) ? arow : (N - 1)) * 128;
    v16bf a_frag[4];
    #pragma unroll
    for (int s = 0; s < 4; ++s) {
        #pragma unroll
        for (int i = 0; i < 16; ++i) {
            int K = s * 32 + i + 16 * laneHi;
            a_frag[s][i] = (__bf16)xrow[K];
        }
    }

    // ---- loop over 8 output column tiles ----
    #pragma unroll 2
    for (int ot = 0; ot < 8; ++ot) {
        v8f c = {0.f, 0.f, 0.f, 0.f, 0.f, 0.f, 0.f, 0.f};
        #pragma unroll
        for (int s = 0; s < 4; ++s) {
            // B (32x16), same contiguous k order as A: lane col = lane&15; element i -> K = i + 16*laneHi
            const __bf16* bp = &Wlds[(ot * 16 + lane16) * 128 + s * 32 + 16 * laneHi];
            v16bf b;
            #pragma unroll
            for (int i = 0; i < 16; ++i) b[i] = bp[i];
            c = __builtin_amdgcn_wmma_f32_16x16x32_bf16(
                    false, a_frag[s], false, b, (short)0, c, false, false);
        }
        // D layout: element r -> (M = r + 8*laneHi, Ncol = lane&15)
        float* trow = th + (size_t)(row0 + 8 * laneHi) * 128 + ot * 16 + lane16;
        if (fullWave) {
            #pragma unroll
            for (int r = 0; r < 8; ++r) trow[(size_t)r * 128] = c[r];
        } else {
            #pragma unroll
            for (int r = 0; r < 8; ++r) {
                if (row0 + 8 * laneHi + r < N) trow[(size_t)r * 128] = c[r];
            }
        }
    }
}

// -------------------------------------------------------------------------
// Kernel 2: per-node attention scores  s_src[n] = th[n]·a[0:128], s_dst[n] = th[n]·a[128:256]
// -------------------------------------------------------------------------
__global__ void gat_scores(const float* __restrict__ th, const float* __restrict__ a,
                           float* __restrict__ s_src, float* __restrict__ s_dst, int N)
{
    int n = blockIdx.x * blockDim.x + threadIdx.x;
    if (n >= N) return;
    const float4* t  = (const float4*)(th + (size_t)n * 128);
    const float4* as = (const float4*)a;
    const float4* ad = (const float4*)(a + 128);
    float ss = 0.f, sd = 0.f;
    #pragma unroll 8
    for (int i = 0; i < 32; ++i) {
        float4 tv = t[i], av = as[i], dv = ad[i];
        ss += tv.x * av.x + tv.y * av.y + tv.z * av.z + tv.w * av.w;
        sd += tv.x * dv.x + tv.y * dv.y + tv.z * dv.z + tv.w * dv.w;
    }
    s_src[n] = ss;
    s_dst[n] = sd;
}

// -------------------------------------------------------------------------
// Kernel 3: zero fill (denom and output must be zeroed every call)
// -------------------------------------------------------------------------
__global__ void gat_zero(float* __restrict__ p, long n)
{
    long i = (long)blockIdx.x * blockDim.x + threadIdx.x;
    if (i < n) p[i] = 0.f;
}

// -------------------------------------------------------------------------
// Kernel 4: per-edge  num = exp(leakyrelu(s_src[src]+s_dst[dst])),
//           denom[dst] += num.  Edges [0,E) from edge_index, [E,E+N) self-loops.
// -------------------------------------------------------------------------
__global__ void gat_edge_num(const long long* __restrict__ ei,
                             const float* __restrict__ s_src, const float* __restrict__ s_dst,
                             float* __restrict__ num, float* __restrict__ denom,
                             int E, int N)
{
    int e = blockIdx.x * blockDim.x + threadIdx.x;
    int Etot = E + N;
    if (e >= Etot) return;
    int s, d;
    if (e < E) { s = (int)ei[e]; d = (int)ei[(size_t)E + e]; }
    else       { s = d = e - E; }
    float sc = s_src[s] + s_dst[d];
    sc = (sc >= 0.f) ? sc : NEG_SLOPE * sc;
    float nm = __expf(sc);
    num[e] = nm;
    atomicAdd(&denom[d], nm);
}

// -------------------------------------------------------------------------
// Kernel 5: wave-per-edge weighted scatter:
//           out[dst] += (num[e]/denom[dst]) * th[src]
// lane L handles columns [4L, 4L+4): 512B coalesced gather + 4 f32 atomics.
// -------------------------------------------------------------------------
__global__ __launch_bounds__(256) void gat_aggregate(
    const long long* __restrict__ ei, const float* __restrict__ th,
    const float* __restrict__ num, const float* __restrict__ denom,
    float* __restrict__ out, int E, int N)
{
    long gid  = (long)blockIdx.x * blockDim.x + threadIdx.x;
    long e    = gid >> 5;
    int  lane = (int)(gid & 31);
    long Etot = (long)E + N;
    if (e >= Etot) return;
    int s, d;
    if (e < E) { s = (int)ei[e]; d = (int)ei[(size_t)E + e]; }
    else       { s = d = (int)(e - E); }
    float dn = denom[d];
    dn = (dn == 0.f) ? 1.f : dn;
    float alpha = num[e] / dn;
    float4 t = ((const float4*)(th + (size_t)s * 128))[lane];
    float* o = out + (size_t)d * 128 + lane * 4;
    atomicAdd(o + 0, alpha * t.x);
    atomicAdd(o + 1, alpha * t.y);
    atomicAdd(o + 2, alpha * t.z);
    atomicAdd(o + 3, alpha * t.w);
}

// -------------------------------------------------------------------------
extern "C" void kernel_launch(void* const* d_in, const int* in_sizes, int n_in,
                              void* d_out, int out_size, void* d_ws, size_t ws_size,
                              hipStream_t stream)
{
    const float*     x  = (const float*)d_in[0];
    const long long* ei = (const long long*)d_in[1];   // int64 (2, E)
    const float*     W  = (const float*)d_in[2];
    const float*     a  = (const float*)d_in[3];

    const int N = in_sizes[0] / 128;
    const int E = in_sizes[1] / 2;

    float* ws    = (float*)d_ws;
    float* th    = ws;                          // N*128
    float* s_src = th + (size_t)N * 128;        // N
    float* s_dst = s_src + N;                   // N
    float* denom = s_dst + N;                   // N
    float* num   = denom + N;                   // E+N
    float* out   = (float*)d_out;               // N*128

    // 1) th = x @ W^T (WMMA bf16 -> f32)
    gat_gemm_th<<<(N + 127) / 128, 256, 0, stream>>>(x, W, th, N);

    // 2) per-node scores
    gat_scores<<<(N + 255) / 256, 256, 0, stream>>>(th, a, s_src, s_dst, N);

    // 3) zero denom + output
    gat_zero<<<(N + 255) / 256, 256, 0, stream>>>(denom, (long)N);
    long outn = (long)N * 128;
    gat_zero<<<(int)((outn + 255) / 256), 256, 0, stream>>>(out, outn);

    // 4) edge softmax numerators + denominator scatter
    int Etot = E + N;
    gat_edge_num<<<(Etot + 255) / 256, 256, 0, stream>>>(ei, s_src, s_dst, num, denom, E, N);

    // 5) wave-per-edge weighted aggregation
    long tot = (long)Etot * 32;
    gat_aggregate<<<(int)((tot + 255) / 256), 256, 0, stream>>>(ei, th, num, denom, out, E, N);
}
